// Attention_23476291239959
// MI455X (gfx1250) — matrix operand order repaired and checked by reading, not compiled, not run on hardware
//
#include <hip/hip_runtime.h>

// Problem dims
#define BQ 32
#define QL 512
#define KL 512
#define AD 256
#define HH 8
#define CD 32
#define OD 256

typedef __attribute__((ext_vector_type(16))) __bf16 v16bf;
typedef __attribute__((ext_vector_type(8)))  float  v8f;

union Frag { uint4 u[2]; v16bf v; };

__device__ inline unsigned short f2bf(float f) {
    unsigned u = __float_as_uint(f);
    u += 0x7FFFu + ((u >> 16) & 1u);     // round-to-nearest-even
    return (unsigned short)(u >> 16);
}
__device__ inline unsigned pack2bf(float lo, float hi) {
    return (unsigned)f2bf(lo) | ((unsigned)f2bf(hi) << 16);
}
__device__ inline float bf2f(unsigned short h) {
    return __uint_as_float(((unsigned)h) << 16);
}
__device__ inline v8f zero8() {
    v8f z;
#pragma unroll
    for (int i = 0; i < 8; ++i) z[i] = 0.f;
    return z;
}

// ---------------------------------------------------------------------------
// GEMM kernel: M=16384, K=256, N=256.  Block tile 128(M) x 64(N), 8 waves,
// each wave owns a 16-row stripe and 4 N-tiles of 16.  bf16 WMMA, f32 acc.
// mode 0: q = q_data*query_w * C^-0.5          -> qp   (bf16)
// mode 1: k = m_data*key_w                     -> kp   (bf16)
// mode 2: v = m_data*value_w                   -> vp   (bf16)
// mode 3: gate = sigmoid(q_data*gating_w + gb) -> gatep(bf16)
// mode 4: out = attn_bf*output_w + output_b    -> f32 d_out
// ---------------------------------------------------------------------------
__global__ __launch_bounds__(256) void gemm_bf16_kernel(
    const float* __restrict__ q_data, const float* __restrict__ m_data,
    const unsigned short* __restrict__ attn_bf,
    const float* __restrict__ query_w, const float* __restrict__ key_w,
    const float* __restrict__ value_w, const float* __restrict__ gating_w,
    const float* __restrict__ gating_b,
    const float* __restrict__ output_w, const float* __restrict__ output_b,
    unsigned short* __restrict__ qp, unsigned short* __restrict__ kp,
    unsigned short* __restrict__ vp, unsigned short* __restrict__ gatep,
    float* __restrict__ out_f32, int mode_base)
{
    __shared__ __align__(16) unsigned short At[128 * 32];   // A tile, row-major
    __shared__ __align__(16) unsigned short Bt[64 * 32];    // B tile, [n][k]

    const int mode = (mode_base < 0) ? (int)blockIdx.z : mode_base;
    const int tid = threadIdx.x;
    const int w   = tid >> 5;
    const int ln  = tid & 31;
    const int g   = ln >> 4;
    const int n16 = ln & 15;

    const int rowBase = blockIdx.x * 128;
    const int n0      = blockIdx.y * 64;

    const float* Asrc = nullptr;
    const float* W    = nullptr;
    if      (mode == 0) { Asrc = q_data; W = query_w;  }
    else if (mode == 1) { Asrc = m_data; W = key_w;    }
    else if (mode == 2) { Asrc = m_data; W = value_w;  }
    else if (mode == 3) { Asrc = q_data; W = gating_w; }
    else                { W = output_w; }

    v8f acc[4];
#pragma unroll
    for (int t = 0; t < 4; ++t) acc[t] = zero8();

    for (int kk = 0; kk < 8; ++kk) {
        const int k0 = kk * 32;
        __syncthreads();
        if (mode == 4) {
            // bf16 source: copy 128x32 tile as uint4 (8 bf16 each), 2 per thread
#pragma unroll
            for (int i = 0; i < 2; ++i) {
                const int gi  = tid + 256 * i;          // group of 8 elems
                const int r   = gi >> 2, off = (gi & 3) * 8;
                ((uint4*)At)[gi] =
                    *(const uint4*)(attn_bf + (size_t)(rowBase + r) * 256 + k0 + off);
            }
        } else {
            // f32 source: two float4 loads -> packed uint4 (8 bf16) store
#pragma unroll
            for (int i = 0; i < 2; ++i) {
                const int gi  = tid + 256 * i;
                const int r   = gi >> 2, off = (gi & 3) * 8;
                const float4* src = (const float4*)(Asrc + (size_t)(rowBase + r) * 256 + k0 + off);
                const float4 a0 = src[0], a1 = src[1];
                uint4 pk;
                pk.x = pack2bf(a0.x, a0.y); pk.y = pack2bf(a0.z, a0.w);
                pk.z = pack2bf(a1.x, a1.y); pk.w = pack2bf(a1.z, a1.w);
                ((uint4*)At)[gi] = pk;
            }
        }
        // B tile 32x64 -> transposed [n][k]
#pragma unroll
        for (int i = 0; i < 8; ++i) {
            const int idx = tid + 256 * i;
            const int r = idx >> 6, c = idx & 63;       // r = k, c = n
            Bt[c * 32 + r] = f2bf(W[(size_t)(k0 + r) * 256 + n0 + c]);
        }
        __syncthreads();

        // A fragment (ISA 16-bit A 16x32 layout): lane m=n16, K chunks by g
        Frag a;
        const uint4* ar = (const uint4*)&At[(w * 16 + n16) * 32];
        a.u[0] = ar[g];          // K = 8g .. 8g+7
        a.u[1] = ar[2 + g];      // K = 16+8g .. 16+8g+7

        // load all four B fragments first so waits can overlap
        Frag bfr[4];
#pragma unroll
        for (int t = 0; t < 4; ++t) {
            const uint4* br = (const uint4*)&Bt[(t * 16 + n16) * 32];
            bfr[t].u[0] = br[g];
            bfr[t].u[1] = br[2 + g];
        }
#pragma unroll
        for (int t = 0; t < 4; ++t)
            acc[t] = __builtin_amdgcn_wmma_f32_16x16x32_bf16(
                false, a.v, false, bfr[t].v, (short)0, acc[t], false, false);
    }

    // Epilogue: C/D layout -> row = rowW + r, col = colW + 16t.
    // Mode branch hoisted out of the loops; loops fully unrolled so acc[t][r]
    // is a fixed VGPR (no v_movrels).
    const int rowW = rowBase + w * 16 + 8 * g;
    const int colW = n0 + n16;
    if (mode == 0) {
#pragma unroll
        for (int t = 0; t < 4; ++t)
#pragma unroll
            for (int r = 0; r < 8; ++r)
                qp[(size_t)(rowW + r) * 256 + colW + 16 * t] =
                    f2bf(acc[t][r] * 0.17677669529663687f);     // 32^-0.5
    } else if (mode == 1) {
#pragma unroll
        for (int t = 0; t < 4; ++t)
#pragma unroll
            for (int r = 0; r < 8; ++r)
                kp[(size_t)(rowW + r) * 256 + colW + 16 * t] = f2bf(acc[t][r]);
    } else if (mode == 2) {
#pragma unroll
        for (int t = 0; t < 4; ++t)
#pragma unroll
            for (int r = 0; r < 8; ++r)
                vp[(size_t)(rowW + r) * 256 + colW + 16 * t] = f2bf(acc[t][r]);
    } else if (mode == 3) {
#pragma unroll
        for (int t = 0; t < 4; ++t) {
            const float gbv = gating_b[colW + 16 * t];
#pragma unroll
            for (int r = 0; r < 8; ++r)
                gatep[(size_t)(rowW + r) * 256 + colW + 16 * t] =
                    f2bf(1.f / (1.f + __expf(-(acc[t][r] + gbv))));
        }
    } else {
#pragma unroll
        for (int t = 0; t < 4; ++t) {
            const float obv = output_b[colW + 16 * t];
#pragma unroll
            for (int r = 0; r < 8; ++r)
                out_f32[(size_t)(rowW + r) * 256 + colW + 16 * t] = acc[t][r] + obv;
        }
    }
}

// ---------------------------------------------------------------------------
// Flash attention: one block per (b, h, 128-q-tile); 8 waves x 16 q-rows.
// Streams 512 keys in 32-key chunks with online softmax; never materializes
// the 268 MB logits tensor.  batched_bias is the roofline floor -> prefetch
// the next chunk's rows (global_prefetch_b8).
// ---------------------------------------------------------------------------
__global__ __launch_bounds__(256) void attn_flash_kernel(
    const unsigned short* __restrict__ qp, const unsigned short* __restrict__ kp,
    const unsigned short* __restrict__ vp, const unsigned short* __restrict__ gatep,
    const float* __restrict__ bias, const float* __restrict__ nb,
    const float* __restrict__ bb, unsigned short* __restrict__ attn_out)
{
    __shared__ __align__(16) unsigned short Kt[32 * 32];        // [key][c]
    __shared__ __align__(16) unsigned short Vt[32 * 32];        // [c][key]
    __shared__ __align__(16) unsigned short Pb[8 * 16 * 32];    // per-wave P

    const int tid = threadIdx.x;
    const int w = tid >> 5, ln = tid & 31, g = ln >> 4, n = ln & 15;
    const int qt = blockIdx.x;   // q tile (0..3)
    const int h  = blockIdx.y;
    const int b  = blockIdx.z;

    const int qw0 = qt * 128 + w * 16;

    // Persistent Q fragment (A layout), loaded straight from global bf16
    Frag qa;
    {
        const uint4* qr = (const uint4*)(qp + ((size_t)(b * QL + qw0 + n) * 256 + h * 32));
        qa.u[0] = qr[g];
        qa.u[1] = qr[2 + g];
    }

    v8f o0 = zero8(), o1 = zero8();
    float mrow[8], lrow[8];
#pragma unroll
    for (int r = 0; r < 8; ++r) { mrow[r] = -1e30f; lrow[r] = 0.f; }

    unsigned short* Pw = Pb + w * 512;

    for (int ch = 0; ch < 16; ++ch) {
        const int k0 = ch * 32;
        __syncthreads();
        // K chunk: 32 keys x 32 c, row-major (uint4 = 8 bf16)
        if (tid < 128) {
            int key = tid >> 2, part = tid & 3;
            ((uint4*)Kt)[tid] =
                ((const uint4*)(kp + ((size_t)(b * KL + k0 + key) * 256 + h * 32)))[part];
        }
        // V chunk transposed: Vt[c][key] so PV B-operand is k-contiguous
#pragma unroll
        for (int i = 0; i < 4; ++i) {
            const int idx = tid + 256 * i;
            const int key = idx >> 5, c = idx & 31;
            Vt[c * 32 + key] = vp[(size_t)(b * KL + k0 + key) * 256 + h * 32 + c];
        }
        // prefetch next chunk of the dominant batched_bias stream
        if (ch + 1 < 16) {
#pragma unroll
            for (int r = 0; r < 8; ++r) {
                const size_t bboN =
                    (((size_t)b * HH + h) * QL + (qw0 + r + 8 * g)) * KL + k0 + 32;
                __builtin_prefetch(&bb[bboN + n], 0, 0);
            }
        }
        __syncthreads();

        // S = Q x K^T : two 16-key WMMAs
        Frag kf0, kf1;
        const uint4* kr0 = (const uint4*)&Kt[n * 32];
        kf0.u[0] = kr0[g]; kf0.u[1] = kr0[2 + g];
        const uint4* kr1 = (const uint4*)&Kt[(16 + n) * 32];
        kf1.u[0] = kr1[g]; kf1.u[1] = kr1[2 + g];

        v8f s0 = zero8(), s1 = zero8();
        s0 = __builtin_amdgcn_wmma_f32_16x16x32_bf16(false, qa.v, false, kf0.v, (short)0, s0, false, false);
        s1 = __builtin_amdgcn_wmma_f32_16x16x32_bf16(false, qa.v, false, kf1.v, (short)0, s1, false, false);

        const float bias0 = bias[(size_t)b * KL + k0 + n];
        const float bias1 = bias[(size_t)b * KL + k0 + 16 + n];

#pragma unroll
        for (int r = 0; r < 8; ++r) {
            const int qabs = qw0 + r + 8 * g;
            const size_t nbo = (size_t)h * QL * KL + (size_t)qabs * KL + k0;
            const size_t bbo = (((size_t)b * HH + h) * QL + qabs) * KL + k0;
            float e0 = s0[r] + bias0 + nb[nbo + n]      + bb[bbo + n];
            float e1 = s1[r] + bias1 + nb[nbo + 16 + n] + bb[bbo + 16 + n];

            // online softmax: reduce across the 16 lanes holding this row
            float t = fmaxf(e0, e1);
#pragma unroll
            for (int msk = 1; msk < 16; msk <<= 1) t = fmaxf(t, __shfl_xor(t, msk, 32));
            const float mnew  = fmaxf(mrow[r], t);
            const float alpha = __expf(mrow[r] - mnew);
            const float p0 = __expf(e0 - mnew);
            const float p1 = __expf(e1 - mnew);
            float rs = p0 + p1;
#pragma unroll
            for (int msk = 1; msk < 16; msk <<= 1) rs += __shfl_xor(rs, msk, 32);
            lrow[r] = lrow[r] * alpha + rs;
            mrow[r] = mnew;
            o0[r] *= alpha; o1[r] *= alpha;

            // spill P (C/D layout) to per-wave LDS for layout conversion
            Pw[(r + 8 * g) * 32 + n]      = f2bf(p0);
            Pw[(r + 8 * g) * 32 + 16 + n] = f2bf(p1);
        }

        // same-wave LDS RAW: drain DS counter before re-reading in A layout
        asm volatile("s_wait_dscnt 0" ::: "memory");

        Frag pf;
        const uint4* pr = (const uint4*)&Pw[n * 32];
        pf.u[0] = pr[g]; pf.u[1] = pr[2 + g];

        Frag vf0, vf1;
        const uint4* vr0 = (const uint4*)&Vt[n * 32];
        vf0.u[0] = vr0[g]; vf0.u[1] = vr0[2 + g];
        const uint4* vr1 = (const uint4*)&Vt[(16 + n) * 32];
        vf1.u[0] = vr1[g]; vf1.u[1] = vr1[2 + g];

        o0 = __builtin_amdgcn_wmma_f32_16x16x32_bf16(false, pf.v, false, vf0.v, (short)0, o0, false, false);
        o1 = __builtin_amdgcn_wmma_f32_16x16x32_bf16(false, pf.v, false, vf1.v, (short)0, o1, false, false);
    }

    // finalize: 1/l, apply sigmoid gate, store bf16 for the output GEMM
#pragma unroll
    for (int r = 0; r < 8; ++r) {
        const int qabs = qw0 + r + 8 * g;
        const float inv = 1.f / lrow[r];
        const size_t base = (size_t)(b * QL + qabs) * 256 + h * 32;
        const float g0 = bf2f(gatep[base + n]);
        const float g1 = bf2f(gatep[base + 16 + n]);
        attn_out[base + n]      = f2bf(o0[r] * inv * g0);
        attn_out[base + 16 + n] = f2bf(o1[r] * inv * g1);
    }
}

// ---------------------------------------------------------------------------
extern "C" void kernel_launch(void* const* d_in, const int* in_sizes, int n_in,
                              void* d_out, int out_size, void* d_ws, size_t ws_size,
                              hipStream_t stream) {
    const float* q_data = (const float*)d_in[0];
    const float* m_data = (const float*)d_in[1];
    const float* bias   = (const float*)d_in[2];
    const float* nbias  = (const float*)d_in[3];
    const float* bbias  = (const float*)d_in[4];
    const float* qw     = (const float*)d_in[5];
    const float* kw     = (const float*)d_in[6];
    const float* vw     = (const float*)d_in[7];
    const float* gw     = (const float*)d_in[8];
    const float* gb     = (const float*)d_in[9];
    const float* ow     = (const float*)d_in[10];
    const float* ob     = (const float*)d_in[11];
    float* out = (float*)d_out;

    // Workspace: 5 bf16 matrices of 16384x256 (q,k,v,gate,attn) = 40 MB
    const size_t MAT = (size_t)BQ * QL * (HH * CD);
    unsigned short* qp    = (unsigned short*)d_ws;
    unsigned short* kp    = qp + MAT;
    unsigned short* vp    = kp + MAT;
    unsigned short* gatep = vp + MAT;
    unsigned short* attnp = gatep + MAT;

    // 1) four projection GEMMs (mode = blockIdx.z)
    gemm_bf16_kernel<<<dim3(128, 4, 4), 256, 0, stream>>>(
        q_data, m_data, attnp, qw, kw, vw, gw, gb, ow, ob,
        qp, kp, vp, gatep, out, -1);

    // 2) flash attention + gate
    attn_flash_kernel<<<dim3(4, HH, BQ), 256, 0, stream>>>(
        qp, kp, vp, gatep, bias, nbias, bbias, attnp);

    // 3) output GEMM + bias
    gemm_bf16_kernel<<<dim3(128, 4, 1), 256, 0, stream>>>(
        q_data, m_data, attnp, qw, kw, vw, gw, gb, ow, ob,
        qp, kp, vp, gatep, out, 4);
}
